// SimpleVisionAttention_21492016349631
// MI455X (gfx1250) — matrix operand order repaired and checked by reading, not compiled, not run on hardware
//
#include <hip/hip_runtime.h>

typedef __attribute__((ext_vector_type(16))) _Float16 v16h;
typedef __attribute__((ext_vector_type(8)))  _Float16 v8h;
typedef __attribute__((ext_vector_type(8)))  float    v8f;

#define NUM_HEADS 16
#define HEAD_DIM  80
#define HEAD_PAD  96
#define DIM       1280
#define SEQ       2048
#define ATTN_SCALE 0.11180339887498948f   /* 80^-0.5 */

#define WMMA_F16(a, b, c) \
  __builtin_amdgcn_wmma_f32_16x16x32_f16(false, (a), false, (b), (short)0, (c), false, false)

static __device__ __forceinline__ v16h cat8(v8h a, v8h b) {
  return __builtin_shufflevector(a, b, 0,1,2,3,4,5,6,7,8,9,10,11,12,13,14,15);
}
static __device__ __forceinline__ v16h ld16(const _Float16* p) {
  return cat8(*(const v8h*)p, *(const v8h*)(p + 8));
}

// ---------------------------------------------------------------------------
// f32 -> f16 elementwise convert (grid-stride)
// ---------------------------------------------------------------------------
__global__ void cvt_f32_f16_kernel(const float* __restrict__ in,
                                   _Float16* __restrict__ out, int n) {
  int i = blockIdx.x * blockDim.x + threadIdx.x;
  int stride = gridDim.x * blockDim.x;
  for (; i < n; i += stride) out[i] = (_Float16)in[i];
}

// ---------------------------------------------------------------------------
// C[M,N] = A[M,K] @ B[N,K]^T + bias[N]    (f16 in, f32 out, WMMA 16x16x32)
//
// Block: 256 threads = 8 waves; block tile = 256 rows x 64 cols.
// Wave w owns rows [bx*256 + 32w, +32) (two 16-row tiles) x all 64 cols.
// The 64x32 B tile per K-step is SHARED by all waves: staged into LDS with
// global_load_async_to_lds_b128 (ASYNCcnt), double-buffered. The last K-step
// is peeled so the steady-state loop prefetches unconditionally (no branches
// in the hot loop). M%256==0, N%64==0, K%64==0 for all our GEMMs.
// ---------------------------------------------------------------------------
__global__ void gemm_bias_wmma_kernel(const _Float16* __restrict__ A,
                                      const _Float16* __restrict__ B,
                                      const float* __restrict__ bias,
                                      float* __restrict__ C,
                                      int M, int N, int K) {
  __shared__ alignas(16) _Float16 Blds[2][64 * 32];

  const int tid  = threadIdx.x;
  const int lane = tid & 31;
  const int wave = tid >> 5;
  const int half = lane >> 4;     // 0: lanes 0-15, 1: lanes 16-31
  const int l16  = lane & 15;
  const int mBase = blockIdx.x * 256 + wave * 32;
  const int nBase = blockIdx.y * 64;

  // async staging map: thread t -> B row (out col) t>>2, 16B chunk t&3
  const int brow = tid >> 2;      // 0..63
  const int bchk = tid & 3;       // 0..3
  const _Float16* Bsrc = B + (size_t)(nBase + brow) * K + bchk * 8;
  const uint32_t ldsOff0 = (uint32_t)(uintptr_t)&Blds[0][brow * 32 + bchk * 8];
  const uint32_t ldsOff1 = (uint32_t)(uintptr_t)&Blds[1][brow * 32 + bchk * 8];

  v8f acc[2][4];
#pragma unroll
  for (int rt = 0; rt < 2; ++rt)
#pragma unroll
    for (int t = 0; t < 4; ++t) acc[rt][t] = (v8f){};

  const _Float16* Arow0 = A + (size_t)(mBase + l16) * K;
  const _Float16* Arow1 = Arow0 + (size_t)16 * K;

  // one K-step body: consume buffer `cur` for K offset k0
  auto step = [&](int ks, int cur) {
    const int k0 = ks * 32;
    // A fragments: lane = row; halves K {k0+8h..+7, k0+16+8h..+7}
    v16h a0 = cat8(*(const v8h*)(Arow0 + k0 + half * 8),
                   *(const v8h*)(Arow0 + k0 + 16 + half * 8));
    v16h a1 = cat8(*(const v8h*)(Arow1 + k0 + half * 8),
                   *(const v8h*)(Arow1 + k0 + 16 + half * 8));
#pragma unroll
    for (int t = 0; t < 4; ++t) {
      const _Float16* bp = &Blds[cur][(t * 16 + l16) * 32 + half * 8];
      v16h bf = cat8(*(const v8h*)bp, *(const v8h*)(bp + 16));
      acc[0][t] = WMMA_F16(a0, bf, acc[0][t]);
      acc[1][t] = WMMA_F16(a1, bf, acc[1][t]);
    }
  };

  // preload K-step 0 into buffer 0
  asm volatile("global_load_async_to_lds_b128 %0, %1, off"
               :: "v"(ldsOff0), "v"(Bsrc) : "memory");

  const int nk = K / 32;
  // steady state: unconditional next-tile prefetch, wait for current only
  for (int ks = 0; ks < nk - 1; ++ks) {
    const int cur = ks & 1;
    const uint32_t nxtOff = (cur == 0) ? ldsOff1 : ldsOff0;
    asm volatile("global_load_async_to_lds_b128 %0, %1, off"
                 :: "v"(nxtOff), "v"(Bsrc + (ks + 1) * 32) : "memory");
    asm volatile("s_wait_asynccnt 0x1" ::: "memory");  // cur done, nxt in flight
    __syncthreads();          // cur B tile visible to all waves
    step(ks, cur);
    __syncthreads();          // all reads of cur done before it is refilled
  }
  // peeled last K-step
  asm volatile("s_wait_asynccnt 0x0" ::: "memory");
  __syncthreads();
  step(nk - 1, (nk - 1) & 1);

  const float bb0 = bias[nBase + l16];
  const float bb1 = bias[nBase + 16 + l16];
  const float bb2 = bias[nBase + 32 + l16];
  const float bb3 = bias[nBase + 48 + l16];
#pragma unroll
  for (int rt = 0; rt < 2; ++rt) {
#pragma unroll
    for (int r = 0; r < 8; ++r) {
      const int row = mBase + rt * 16 + r + half * 8;
      float* Crow = C + (size_t)row * N + nBase + l16;
      Crow[0]  = acc[rt][0][r] + bb0;
      Crow[16] = acc[rt][1][r] + bb1;
      Crow[32] = acc[rt][2][r] + bb2;
      Crow[48] = acc[rt][3][r] + bb3;
    }
  }
}

// ---------------------------------------------------------------------------
// RoPE + repack: qkv f32 [s][3][h][80] ->
//   Q16/K16: [h][s][96] f16 (rope applied, padded 80->96 with zeros)
//   Vt16:    [h][80][2048] f16 (transposed for P@V B-fragments)
// ---------------------------------------------------------------------------
__global__ void rope_pack_kernel(const float* __restrict__ qkv,
                                 const float* __restrict__ cosb,
                                 const float* __restrict__ sinb,
                                 _Float16* __restrict__ Q16,
                                 _Float16* __restrict__ K16,
                                 _Float16* __restrict__ Vt16) {
  int idx = blockIdx.x * blockDim.x + threadIdx.x;
  const int total = SEQ * NUM_HEADS * HEAD_PAD;
  if (idx >= total) return;
  const int d = idx % HEAD_PAD;
  const int h = (idx / HEAD_PAD) % NUM_HEADS;
  const int s = idx / (HEAD_PAD * NUM_HEADS);
  const size_t qoff = (size_t)h * SEQ * HEAD_PAD + (size_t)s * HEAD_PAD + d;
  if (d >= HEAD_DIM) { Q16[qoff] = (_Float16)0.f; K16[qoff] = (_Float16)0.f; return; }

  const size_t rowq = (size_t)s * (3 * DIM);
  const int ch = h * HEAD_DIM + d;
  const float c  = cosb[s * HEAD_DIM + d];
  const float sn = sinb[s * HEAD_DIM + d];
  const int dr   = (d < HEAD_DIM / 2) ? d + HEAD_DIM / 2 : d - HEAD_DIM / 2;
  const float sg = (d < HEAD_DIM / 2) ? -1.f : 1.f;

  const float q  = qkv[rowq + ch];
  const float k  = qkv[rowq + DIM + ch];
  const float v  = qkv[rowq + 2 * DIM + ch];
  const float qr = sg * qkv[rowq + h * HEAD_DIM + dr];
  const float kr = sg * qkv[rowq + DIM + h * HEAD_DIM + dr];

  Q16[qoff] = (_Float16)(q * c + qr * sn);
  K16[qoff] = (_Float16)(k * c + kr * sn);
  Vt16[(size_t)h * HEAD_DIM * SEQ + (size_t)d * SEQ + s] = (_Float16)v;
}

// ---------------------------------------------------------------------------
// Flash attention. One wave = one (head, 16-query block). 128 thr = 4 waves.
// Scores: S[16,32] = Q(16x96) @ K^T via 6 WMMAs; online softmax; P staged
// through LDS (C-layout -> A-fragment re-layout); O += P @ V via 5 WMMAs.
// ---------------------------------------------------------------------------
__global__ void attn_wmma_kernel(const _Float16* __restrict__ Q16,
                                 const _Float16* __restrict__ K16,
                                 const _Float16* __restrict__ Vt16,
                                 _Float16* __restrict__ AO16) {
  __shared__ alignas(16) _Float16 Plds[4][16 * 32];
  const int lane = threadIdx.x & 31;
  const int wave = threadIdx.x >> 5;
  const int half = lane >> 4;
  const int l16  = lane & 15;
  const int task = blockIdx.x * 4 + wave;
  const int h  = task >> 7;     // / (SEQ/16)
  const int qb = task & 127;

  const _Float16* Qh = Q16 + (size_t)h * SEQ * HEAD_PAD;
  const _Float16* Kh = K16 + (size_t)h * SEQ * HEAD_PAD;
  const _Float16* Vh = Vt16 + (size_t)h * HEAD_DIM * SEQ;

  // Q fragments for the 3 K-chunks of the padded head dim, loaded once.
  v16h qf[3];
  const _Float16* qrow = Qh + (size_t)(qb * 16 + l16) * HEAD_PAD;
#pragma unroll
  for (int c = 0; c < 3; ++c)
    qf[c] = cat8(*(const v8h*)(qrow + c * 32 + half * 8),
                 *(const v8h*)(qrow + c * 32 + 16 + half * 8));

  float m_i[8], l_i[8];
  v8f o[5];
#pragma unroll
  for (int r = 0; r < 8; ++r) { m_i[r] = -1e30f; l_i[r] = 0.f; }
#pragma unroll
  for (int dt = 0; dt < 5; ++dt) o[dt] = (v8f){};

  for (int t0 = 0; t0 < SEQ; t0 += 32) {
    v8f S0 = {}, S1 = {};
#pragma unroll
    for (int c = 0; c < 3; ++c) {
      const _Float16* kp0 = Kh + (size_t)(t0 + l16) * HEAD_PAD + c * 32 + half * 8;
      const _Float16* kp1 = kp0 + (size_t)16 * HEAD_PAD;
      v16h kf0 = cat8(*(const v8h*)kp0, *(const v8h*)(kp0 + 16));
      v16h kf1 = cat8(*(const v8h*)kp1, *(const v8h*)(kp1 + 16));
      S0 = WMMA_F16(qf[c], kf0, S0);
      S1 = WMMA_F16(qf[c], kf1, S1);
    }
    // Online softmax over the 32 new columns, row by row (row = r + 8*half).
#pragma unroll
    for (int r = 0; r < 8; ++r) {
      float s0 = S0[r] * ATTN_SCALE;
      float s1 = S1[r] * ATTN_SCALE;
      float mx = fmaxf(s0, s1);
#pragma unroll
      for (int off = 1; off < 16; off <<= 1)
        mx = fmaxf(mx, __shfl_xor(mx, off, 32));
      const float mnew  = fmaxf(m_i[r], mx);
      const float alpha = __expf(m_i[r] - mnew);
      const float p0 = __expf(s0 - mnew);
      const float p1 = __expf(s1 - mnew);
      float rs = p0 + p1;
#pragma unroll
      for (int off = 1; off < 16; off <<= 1)
        rs += __shfl_xor(rs, off, 32);
      l_i[r] = l_i[r] * alpha + rs;
      m_i[r] = mnew;
#pragma unroll
      for (int dt = 0; dt < 5; ++dt) o[dt][r] *= alpha;
      Plds[wave][(r + half * 8) * 32 + l16]      = (_Float16)p0;
      Plds[wave][(r + half * 8) * 32 + 16 + l16] = (_Float16)p1;
    }
    // Reload P as an A-fragment (16x32) from LDS.
    const _Float16* pr = &Plds[wave][l16 * 32];
    v16h pf = cat8(*(const v8h*)(pr + half * 8),
                   *(const v8h*)(pr + 16 + half * 8));
#pragma unroll
    for (int dt = 0; dt < 5; ++dt) {
      const _Float16* vp = Vh + (size_t)(dt * 16 + l16) * SEQ + t0 + half * 8;
      v16h vf = cat8(*(const v8h*)vp, *(const v8h*)(vp + 16));
      o[dt] = WMMA_F16(pf, vf, o[dt]);
    }
  }

  // Normalize and emit [s][h*80+d] f16 for the projection GEMM.
#pragma unroll
  for (int dt = 0; dt < 5; ++dt) {
#pragma unroll
    for (int r = 0; r < 8; ++r) {
      const int srow = qb * 16 + r + half * 8;
      AO16[(size_t)srow * DIM + h * HEAD_DIM + dt * 16 + l16] =
          (_Float16)(o[dt][r] / l_i[r]);
    }
  }
}

// ---------------------------------------------------------------------------
// Host-side launch
// ---------------------------------------------------------------------------
extern "C" void kernel_launch(void* const* d_in, const int* in_sizes, int n_in,
                              void* d_out, int out_size, void* d_ws, size_t ws_size,
                              hipStream_t stream) {
  const float* hidden = (const float*)d_in[0];
  /* d_in[1] = cu_seqlens (single sequence, unused) */
  const float* cosb   = (const float*)d_in[2];
  const float* sinb   = (const float*)d_in[3];
  const float* w_qkv  = (const float*)d_in[4];
  const float* b_qkv  = (const float*)d_in[5];
  const float* w_proj = (const float*)d_in[6];
  const float* b_proj = (const float*)d_in[7];
  float* out = (float*)d_out;

  char* w = (char*)d_ws;
  const size_t off_X16    = 0;                                           // 2048*1280 f16
  const size_t off_Wqkv16 = off_X16    + (size_t)SEQ * DIM * 2;          // 3840*1280 f16
  const size_t off_Wproj  = off_Wqkv16 + (size_t)3 * DIM * DIM * 2;      // 1280*1280 f16
  const size_t off_QKV    = off_Wproj  + (size_t)DIM * DIM * 2;          // 2048*3840 f32
  const size_t off_Q16    = off_QKV    + (size_t)SEQ * 3 * DIM * 4;      // 16*2048*96 f16
  const size_t off_K16    = off_Q16    + (size_t)NUM_HEADS * SEQ * HEAD_PAD * 2;
  const size_t off_Vt16   = off_K16    + (size_t)NUM_HEADS * SEQ * HEAD_PAD * 2;
  const size_t off_AO16   = off_Vt16   + (size_t)NUM_HEADS * HEAD_DIM * SEQ * 2;

  _Float16* X16    = (_Float16*)(w + off_X16);
  _Float16* Wqkv16 = (_Float16*)(w + off_Wqkv16);
  _Float16* Wproj16= (_Float16*)(w + off_Wproj);
  float*    QKV    = (float*)   (w + off_QKV);
  _Float16* Q16    = (_Float16*)(w + off_Q16);
  _Float16* K16    = (_Float16*)(w + off_K16);
  _Float16* Vt16   = (_Float16*)(w + off_Vt16);
  _Float16* AO16   = (_Float16*)(w + off_AO16);

  // 1) precision conversion (layouts already B-fragment friendly: [n][k])
  cvt_f32_f16_kernel<<<2048, 256, 0, stream>>>(hidden, X16, SEQ * DIM);
  cvt_f32_f16_kernel<<<2048, 256, 0, stream>>>(w_qkv, Wqkv16, 3 * DIM * DIM);
  cvt_f32_f16_kernel<<<2048, 256, 0, stream>>>(w_proj, Wproj16, DIM * DIM);

  // 2) QKV = X @ Wqkv^T + b_qkv   (block tile 256x64, async B staging)
  dim3 g1(SEQ / 256, (3 * DIM) / 64);
  gemm_bias_wmma_kernel<<<g1, 256, 0, stream>>>(X16, Wqkv16, b_qkv, QKV,
                                                SEQ, 3 * DIM, DIM);

  // 3) RoPE + pack q/k (padded) + v (transposed)
  const int rope_total = SEQ * NUM_HEADS * HEAD_PAD;
  rope_pack_kernel<<<(rope_total + 255) / 256, 256, 0, stream>>>(
      QKV, cosb, sinb, Q16, K16, Vt16);

  // 4) flash attention: 16 heads * 128 q-blocks, 4 waves/block
  attn_wmma_kernel<<<(NUM_HEADS * (SEQ / 16)) / 4, 128, 0, stream>>>(
      Q16, K16, Vt16, AO16);

  // 5) out = AO @ Wproj^T + b_proj  (f32 result straight to d_out)
  dim3 g2(SEQ / 256, DIM / 64);
  gemm_bias_wmma_kernel<<<g2, 256, 0, stream>>>(AO16, Wproj16, b_proj, out,
                                                SEQ, DIM, DIM);
}
